// SpatialAxialAttention_29764123361422
// MI455X (gfx1250) — matrix operand order repaired and checked by reading, not compiled, not run on hardware
//
#include <hip/hip_runtime.h>
#include <hip/hip_bf16.h>

// ---------------- types & helpers -----------------------------------------
typedef __attribute__((ext_vector_type(16))) __bf16       v16bf;
typedef __attribute__((ext_vector_type(8)))  float        v8f;
typedef __attribute__((ext_vector_type(8)))  unsigned int v8u;
typedef __attribute__((ext_vector_type(4)))  int          i4v;

#define BT      8
#define SEQ     1024      // H*W
#define HEADS   16
#define DH      64
#define DMODEL  1024
#define N3      3072
#define ROWS    8192      // BT*SEQ

// ---- CDNA5 async global->LDS copies (ASYNCcnt-tracked), guarded -----------
#if defined(__has_builtin)
#if __has_builtin(__builtin_amdgcn_global_load_async_to_lds_b128) && \
    __has_builtin(__builtin_amdgcn_s_wait_asynccnt)
#define HAVE_ASYNC_LDS 1
#endif
#endif

#ifdef HAVE_ASYNC_LDS
typedef __attribute__((address_space(1))) i4v g_i4;   // global int4
typedef __attribute__((address_space(3))) i4v l_i4;   // LDS int4
__device__ __forceinline__ void async_b128(const void* g, void* l) {
  __builtin_amdgcn_global_load_async_to_lds_b128((g_i4*)g, (l_i4*)l, 0, 0);
}
#define ASYNC_WAIT(n) __builtin_amdgcn_s_wait_asynccnt(n)
#else
#define ASYNC_WAIT(n)
#endif

__device__ __forceinline__ unsigned short f2bf(float f) {
  unsigned int u = __builtin_bit_cast(unsigned int, f);
  u += 0x7FFFu + ((u >> 16) & 1u);              // round-to-nearest-even
  return (unsigned short)(u >> 16);
}
__device__ __forceinline__ float bf2f(unsigned short h) {
  unsigned int u = ((unsigned int)h) << 16;
  return __builtin_bit_cast(float, u);
}
__device__ __forceinline__ v8f wmma_bf16(const v8u a, const v8u b, v8f c) {
  return __builtin_amdgcn_wmma_f32_16x16x32_bf16(
      false, __builtin_bit_cast(v16bf, a),
      false, __builtin_bit_cast(v16bf, b),
      (short)0, c, false, false);
}

// ---------------- fp32 -> bf16 convert -------------------------------------
__global__ void k_f32_to_bf16(const float* __restrict__ src,
                              unsigned short* __restrict__ dst, int n) {
  int i = blockIdx.x * blockDim.x + threadIdx.x;
  if (i < n) dst[i] = f2bf(src[i]);
}

// ---------------- bf16 GEMM: C[M,N] = A[M,K] * B[K,N] ----------------------
// 128x128 tile per workgroup (8 waves, wave32); BK = 32; double-buffered.
template <bool BF16_OUT>
__global__ __launch_bounds__(256) void k_gemm_bf16(
    const unsigned short* __restrict__ A, const unsigned short* __restrict__ B,
    void* __restrict__ Cptr, const float* __restrict__ bias,
    int M, int N, int K) {
  const int tid  = threadIdx.x;
  const int wave = tid >> 5;
  const int lane = tid & 31;
  const int g    = lane >> 4;
  const int ln   = lane & 15;

  const int n0 = blockIdx.x * 128;
  const int m0 = blockIdx.y * 128;

  __shared__ unsigned short As[2][128 * 32];   // [m][k]  (K contiguous)
  __shared__ unsigned short Bs[2][128 * 32];   // [n][k]  (transposed)

  // staging coordinates (constant per thread)
  const int arow = tid >> 2;                // A: 0..63 (+64 on 2nd copy)
  const int aq   = tid & 3;                 // A: uint4 within 32-elem K slab
  const int bk0  = tid >> 6;                // B: 0..3  (+4i)
  const int bnc  = tid & 63;                // B: dword column within 128-wide N

  auto stageA = [&](int k0, int buf) {
    const uint4* ag = (const uint4*)(A + (size_t)(m0 + arow) * K + k0) + aq;
    uint4* al = (uint4*)(As[buf]) + arow * 4 + aq;
    __builtin_prefetch((const char*)ag + 128, 0, 1);
#ifdef HAVE_ASYNC_LDS
    async_b128(ag, al);
    async_b128(ag + (size_t)64 * (K >> 3), al + 64 * 4);
#else
    uint4 t0 = ag[0];
    uint4 t1 = ag[(size_t)64 * (K >> 3)];
    al[0]      = t0;
    al[64 * 4] = t1;
#endif
  };
  auto loadB = [&](int k0, unsigned int* tb) {
    const unsigned int* bg =
        (const unsigned int*)(B + (size_t)(k0 + bk0) * N + n0) + bnc;
#pragma unroll
    for (int i = 0; i < 8; ++i) tb[i] = bg[(size_t)i * 4 * (N >> 1)];
  };
  auto storeB = [&](int buf, const unsigned int* tb) {
#pragma unroll
    for (int i = 0; i < 8; ++i) {
      int kk = bk0 + 4 * i;
      Bs[buf][(2 * bnc) * 32 + kk]     = (unsigned short)(tb[i] & 0xffffu);
      Bs[buf][(2 * bnc + 1) * 32 + kk] = (unsigned short)(tb[i] >> 16);
    }
  };

  v8f zero8 = {0.f, 0.f, 0.f, 0.f, 0.f, 0.f, 0.f, 0.f};
  v8f acc[8];
#pragma unroll
  for (int t = 0; t < 8; ++t) acc[t] = zero8;

  // prologue: tile 0 into buffer 0
  {
    unsigned int tb[8];
    stageA(0, 0);
    loadB(0, tb);
    storeB(0, tb);
  }

  const int nk = K >> 5;
#pragma unroll 1
  for (int t = 0; t < nk; ++t) {
    const int  cur = t & 1, nxt = cur ^ 1;
    const bool hasNext = (t + 1) < nk;
    unsigned int tb2[8];
    if (hasNext) {                       // kick next tile before computing
      stageA((t + 1) * 32, nxt);
      loadB((t + 1) * 32, tb2);
    }
    if (hasNext) { ASYNC_WAIT(2); } else { ASYNC_WAIT(0); }
    __syncthreads();                     // As[cur], Bs[cur] ready

    // ---- A fragment (16x32, §7.12.2 16-bit A layout) ----
    v8u af;
    {
      const unsigned int* Alu =
          (const unsigned int*)(As[cur] + (wave * 16 + ln) * 32);
#pragma unroll
      for (int v = 0; v < 8; ++v) {
        int kb = ((v < 4) ? 0 : 16) + g * 8 + (v & 3) * 2;
        af[v] = Alu[kb >> 1];
      }
    }
    // ---- 8 WMMAs across the 128-wide N strip ----
#pragma unroll
    for (int nt = 0; nt < 8; ++nt) {
      v8u bfrag;
      const unsigned int* Blu =
          (const unsigned int*)(Bs[cur] + (nt * 16 + ln) * 32);
#pragma unroll
      for (int v = 0; v < 8; ++v) {
        int kk = ((v < 4) ? 0 : 16) + g * 8 + (v & 3) * 2;
        bfrag[v] = Blu[kk >> 1];
      }
      acc[nt] = wmma_bf16(af, bfrag, acc[nt]);
    }
    if (hasNext) storeB(nxt, tb2);       // commit next B after compute
    __syncthreads();
  }

  // ---- epilogue (branch-free, pointer-strided) ----
  if constexpr (BF16_OUT) {
    unsigned short* C = (unsigned short*)Cptr +
                        (size_t)(m0 + wave * 16 + g * 8) * N + n0 + ln;
#pragma unroll
    for (int r = 0; r < 8; ++r) {
#pragma unroll
      for (int nt = 0; nt < 8; ++nt) C[nt * 16] = f2bf(acc[nt][r]);
      C += N;
    }
  } else {
    float bv[8];
#pragma unroll
    for (int nt = 0; nt < 8; ++nt) bv[nt] = bias[n0 + nt * 16 + ln];
    float* C = (float*)Cptr +
               (size_t)(m0 + wave * 16 + g * 8) * N + n0 + ln;
#pragma unroll
    for (int r = 0; r < 8; ++r) {
#pragma unroll
      for (int nt = 0; nt < 8; ++nt) C[nt * 16] = acc[nt][r] + bv[nt];
      C += N;
    }
  }
}

// ---------------- RoPE on q and k halves of qkv (in place, bf16) -----------
__global__ void k_rope(unsigned short* __restrict__ qkv, int npairs) {
  int i = blockIdx.x * blockDim.x + threadIdx.x;
  if (i >= npairs) return;
  int row = i >> 10;            // 1024 pairs per row (q+k = 2048 cols)
  int col = (i & 1023) * 2;     // 0..2046, <1024 => q else k
  int c   = col & 63;           // channel within head
  int h   = (row >> 5) & 31;
  int w   = row & 31;
  float p    = (c < 32) ? (-1.f + 2.f * h / 31.f) : (-1.f + 2.f * w / 31.f);
  int   j    = (c & 31) >> 1;
  float base = (1.f + (float)j * (127.f / 15.f)) * 3.14159265358979f;
  float f    = p * base;
  float cf = __cosf(f), sf = __sinf(f);
  size_t off = (size_t)row * N3 + col;
  float x1 = bf2f(qkv[off]), x2 = bf2f(qkv[off + 1]);
  qkv[off]     = f2bf(x1 * cf - x2 * sf);
  qkv[off + 1] = f2bf(x2 * cf + x1 * sf);
}

// ---------------- flash attention: per (q-block, head, bt) -----------------
// Double-buffered K/V staging overlapped with the WMMA stream.
__global__ __launch_bounds__(256) void k_attn(
    const unsigned short* __restrict__ qkv, unsigned short* __restrict__ Obf) {
  const int tid  = threadIdx.x;
  const int wave = tid >> 5;
  const int lane = tid & 31;
  const int g    = lane >> 4;
  const int ln   = lane & 15;

  const int qblk = blockIdx.x;          // 0..7
  const int head = blockIdx.y;          // 0..15
  const int bt   = blockIdx.z;          // 0..7
  const int qrow0 = qblk * 128 + wave * 16;
  const size_t rowbase = (size_t)bt * SEQ;

  __shared__ unsigned short Kb[2][64 * 64];  // [key][dh]
  __shared__ unsigned short Vb[2][64 * 64];  // [dh][key] (transposed)
  __shared__ unsigned short Ps[8][16 * 64];  // per-wave P staging

  // staging coordinates (constant per thread)
  const int kq    = tid & 7;            // K: uint4 within 64-elem dh row
  const int key0k = tid >> 3;           // K: 0..31 (+32 on 2nd copy)
  const int vc    = tid & 31;           // V: dword within dh row
  const int key0v = tid >> 5;           // V: 0..7 (+8i)

  const unsigned short* kg_base = qkv + rowbase * N3 + DMODEL + head * DH;
  const unsigned short* vg_base = qkv + rowbase * N3 + 2 * DMODEL + head * DH;

  auto stageK = [&](int kb0, int buf) {
    const uint4* kg = (const uint4*)(kg_base + (size_t)(kb0 + key0k) * N3) + kq;
    uint4* kl = (uint4*)(Kb[buf]) + key0k * 8 + kq;
#ifdef HAVE_ASYNC_LDS
    async_b128(kg, kl);
    async_b128(kg + (size_t)32 * (N3 >> 3), kl + 32 * 8);
#else
    uint4 t0 = kg[0];
    uint4 t1 = kg[(size_t)32 * (N3 >> 3)];
    kl[0]      = t0;
    kl[32 * 8] = t1;
#endif
  };
  auto loadV = [&](int kb0, unsigned int* tv) {
    const unsigned int* vg =
        (const unsigned int*)(vg_base + (size_t)(kb0 + key0v) * N3) + vc;
#pragma unroll
    for (int i = 0; i < 8; ++i) tv[i] = vg[(size_t)i * 8 * (N3 >> 1)];
  };
  auto storeV = [&](int buf, const unsigned int* tv) {
#pragma unroll
    for (int i = 0; i < 8; ++i) {
      int key = key0v + 8 * i;
      Vb[buf][(2 * vc) * 64 + key]     = (unsigned short)(tv[i] & 0xffffu);
      Vb[buf][(2 * vc + 1) * 64 + key] = (unsigned short)(tv[i] >> 16);
    }
  };

  // Q fragments (16x64 -> two 16x32 A-frags), loaded once from global
  v8u qf[2];
  {
    const unsigned short* qrow =
        qkv + (rowbase + qrow0 + ln) * N3 + head * DH;
#pragma unroll
    for (int f = 0; f < 2; ++f)
#pragma unroll
      for (int v = 0; v < 8; ++v) {
        int kb = f * 32 + ((v < 4) ? 0 : 16) + g * 8 + (v & 3) * 2;
        qf[f][v] = *(const unsigned int*)(qrow + kb);
      }
  }

  v8f zero8 = {0.f, 0.f, 0.f, 0.f, 0.f, 0.f, 0.f, 0.f};
  v8f oacc[4];
  float m_i[8], l_i[8];
#pragma unroll
  for (int t = 0; t < 4; ++t) oacc[t] = zero8;
#pragma unroll
  for (int r = 0; r < 8; ++r) { m_i[r] = -1e30f; l_i[r] = 0.f; }
  const float scale = 0.125f;   // 1/sqrt(64)

  // prologue: key block 0 into buffer 0
  {
    unsigned int tv[8];
    stageK(0, 0);
    loadV(0, tv);
    storeV(0, tv);
  }

#pragma unroll 1
  for (int t = 0; t < SEQ / 64; ++t) {
    const int  cur = t & 1, nxt = cur ^ 1;
    const bool hasNext = (t + 1) < SEQ / 64;
    unsigned int tv2[8];
    if (hasNext) {                      // kick next block before computing
      stageK((t + 1) * 64, nxt);
      loadV((t + 1) * 64, tv2);
    }
    if (hasNext) { ASYNC_WAIT(2); } else { ASYNC_WAIT(0); }
    __syncthreads();                    // Kb[cur], Vb[cur] ready

    // ---- S = Q K^T : 4 key sub-tiles x (K=64 -> 2 WMMAs) ----
    v8f sacc[4];
#pragma unroll
    for (int nt = 0; nt < 4; ++nt) {
      v8f s = zero8;
#pragma unroll
      for (int f = 0; f < 2; ++f) {
        v8u bfrag;
#pragma unroll
        for (int v = 0; v < 8; ++v) {
          int kk = f * 32 + ((v < 4) ? 0 : 16) + g * 8 + (v & 3) * 2;
          bfrag[v] =
              *(const unsigned int*)(Kb[cur] + (nt * 16 + ln) * 64 + kk);
        }
        s = wmma_bf16(qf[f], bfrag, s);
      }
      sacc[nt] = s;
    }

    // ---- online softmax (row r+8g; cols across each 16-lane half) ----
#pragma unroll
    for (int r = 0; r < 8; ++r) {
      float mx = fmaxf(fmaxf(sacc[0][r], sacc[1][r]),
                       fmaxf(sacc[2][r], sacc[3][r]));
#pragma unroll
      for (int off = 1; off < 16; off <<= 1)
        mx = fmaxf(mx, __shfl_xor(mx, off, 32));
      mx *= scale;
      float mnew = fmaxf(m_i[r], mx);
      float corr = __expf(m_i[r] - mnew);
      float rs = 0.f;
#pragma unroll
      for (int nt = 0; nt < 4; ++nt) {
        float pv = __expf(sacc[nt][r] * scale - mnew);
        sacc[nt][r] = pv;
        rs += pv;
      }
#pragma unroll
      for (int off = 1; off < 16; off <<= 1)
        rs += __shfl_xor(rs, off, 32);
      l_i[r] = l_i[r] * corr + rs;
      m_i[r] = mnew;
#pragma unroll
      for (int t2 = 0; t2 < 4; ++t2) oacc[t2][r] *= corr;
    }

    // ---- P: C-layout -> LDS -> A-layout (bf16) ----
    unsigned short* pw = Ps[wave];
#pragma unroll
    for (int nt = 0; nt < 4; ++nt)
#pragma unroll
      for (int r = 0; r < 8; ++r)
        pw[(r + 8 * g) * 64 + nt * 16 + ln] = f2bf(sacc[nt][r]);

    v8u pf[2];
    {
      const unsigned short* prow = pw + ln * 64;
#pragma unroll
      for (int f = 0; f < 2; ++f)
#pragma unroll
        for (int v = 0; v < 8; ++v) {
          int kb = f * 32 + ((v < 4) ? 0 : 16) + g * 8 + (v & 3) * 2;
          pf[f][v] = *(const unsigned int*)(prow + kb);
        }
    }

    // ---- O += P V : 4 dh sub-tiles x 2 WMMAs ----
#pragma unroll
    for (int dt = 0; dt < 4; ++dt) {
#pragma unroll
      for (int f = 0; f < 2; ++f) {
        v8u bfrag;
#pragma unroll
        for (int v = 0; v < 8; ++v) {
          int kk = f * 32 + ((v < 4) ? 0 : 16) + g * 8 + (v & 3) * 2;
          bfrag[v] =
              *(const unsigned int*)(Vb[cur] + (dt * 16 + ln) * 64 + kk);
        }
        oacc[dt] = wmma_bf16(pf[f], bfrag, oacc[dt]);
      }
    }
    if (hasNext) storeV(nxt, tv2);      // commit next V after compute
    __syncthreads();
  }

  // ---- epilogue: O / l -> bf16 (pointer-strided) ----
  unsigned short* op =
      Obf + (rowbase + qrow0 + 8 * g) * DMODEL + head * DH + ln;
#pragma unroll
  for (int r = 0; r < 8; ++r) {
    float inv = 1.f / l_i[r];
#pragma unroll
    for (int dt = 0; dt < 4; ++dt) op[dt * 16] = f2bf(oacc[dt][r] * inv);
    op += DMODEL;
  }
}

// ---------------- launch ----------------------------------------------------
extern "C" void kernel_launch(void* const* d_in, const int* in_sizes, int n_in,
                              void* d_out, int out_size, void* d_ws,
                              size_t ws_size, hipStream_t stream) {
  const float* x    = (const float*)d_in[0];
  const float* Wqkv = (const float*)d_in[1];
  const float* Wout = (const float*)d_in[2];
  const float* bout = (const float*)d_in[3];
  float* out = (float*)d_out;

  char* ws = (char*)d_ws;                      // ~92 MB scratch layout
  unsigned short* wqkv_bf = (unsigned short*)(ws);              // 1024*3072
  unsigned short* wout_bf = (unsigned short*)(ws +  6291456);   // 1024*1024
  unsigned short* x_bf    = (unsigned short*)(ws +  8388608);   // 8192*1024
  unsigned short* qkv_bf  = (unsigned short*)(ws + 25165824);   // 8192*3072
  unsigned short* o_bf    = (unsigned short*)(ws + 75497472);   // 8192*1024

  k_f32_to_bf16<<<(3145728 + 255) / 256, 256, 0, stream>>>(Wqkv, wqkv_bf, 3145728);
  k_f32_to_bf16<<<(1048576 + 255) / 256, 256, 0, stream>>>(Wout, wout_bf, 1048576);
  k_f32_to_bf16<<<(8388608 + 255) / 256, 256, 0, stream>>>(x, x_bf, 8388608);

  k_gemm_bf16<true><<<dim3(24, 64), 256, 0, stream>>>(
      x_bf, wqkv_bf, (void*)qkv_bf, nullptr, ROWS, N3, DMODEL);

  k_rope<<<8388608 / 256, 256, 0, stream>>>(qkv_bf, 8388608);

  k_attn<<<dim3(8, HEADS, BT), 256, 0, stream>>>(qkv_bf, o_bf);

  k_gemm_bf16<false><<<dim3(8, 64), 256, 0, stream>>>(
      o_bf, wout_bf, (void*)out, bout, ROWS, DMODEL, DMODEL);
}
